// GPT5Block_86371792323173
// MI455X (gfx1250) — compile-verified, matrix operand order checked by reading, not run on hardware
//
#include <hip/hip_runtime.h>
#include <hip/hip_bf16.h>
#include <stdint.h>

typedef __bf16 bf16;
typedef __attribute__((ext_vector_type(16))) __bf16 v16bf;
typedef __attribute__((ext_vector_type(8)))  float  v8f;

#define D_    512
#define EXP_  16
#define DH_   2048
#define DS2_  1024
#define NTOK  4096

// ---------------- CDNA5 async global->LDS copy helpers ----------------
__device__ __forceinline__ void async_b128(void* ldsp, const void* gp) {
  uint32_t l = (uint32_t)(uintptr_t)ldsp;   // low 32 bits of flat shared addr = LDS offset
  asm volatile("global_load_async_to_lds_b128 %0, %1, off" :: "v"(l), "v"(gp) : "memory");
}
__device__ __forceinline__ void wait_async_le6() { asm volatile("s_wait_asynccnt 0x6" ::: "memory"); }
__device__ __forceinline__ void wait_async_0()  { asm volatile("s_wait_asynccnt 0x0" ::: "memory"); }

// Load one 16-bit WMMA fragment (A 16x32 or B 32x16 layout) from LDS.
// p points at row*40 + kh  (kh = (lane>>4)*8); chunk1 lives at +16 elements.
__device__ __forceinline__ v16bf frag_ld(const bf16* p) {
  union { v16bf v; uint4 u[2]; } t;
  t.u[0] = *(const uint4*)(p);
  t.u[1] = *(const uint4*)(p + 16);
  return t.v;
}

// ---------------- weight transpose + fp32->bf16 convert ----------------
// in: [batch][R][C] f32 row-major   out: [batch][C][R] bf16 row-major (i.e. B^T)
__global__ __launch_bounds__(256) void convT_k(const float* __restrict__ in,
                                               bf16* __restrict__ out, int R, int C) {
  size_t base = (size_t)blockIdx.z * (size_t)R * (size_t)C;
  int o = blockIdx.x * 256 + threadIdx.x;
  if (o >= R * C) return;
  int k = o % R, n = o / R;
  out[base + o] = (bf16)in[base + (size_t)k * C + n];
}

__global__ void zero_counts_k(int* c) { if (threadIdx.x < EXP_) c[threadIdx.x] = 0; }

__global__ void prefix16_k(const int* __restrict__ c, int* __restrict__ o) {
  if (threadIdx.x == 0) { int s = 0; for (int e = 0; e < EXP_; ++e) { o[e] = s; s += c[e]; } }
}

// ---------------- LayerNorm + router + top-2 dispatch (1 wave / token) ----------------
__global__ __launch_bounds__(256) void ln_router_k(
    const float* __restrict__ x, const float* __restrict__ g, const float* __restrict__ b,
    const float* __restrict__ rw, const float* __restrict__ rb,
    bf16* __restrict__ h, int* __restrict__ toklist, float* __restrict__ wlist,
    int* __restrict__ counts)
{
  int token = blockIdx.x * 8 + (threadIdx.x >> 5);
  int lane  = threadIdx.x & 31;
  const float* xr = x + (size_t)token * D_;
  float v[16];
  float sum = 0.f;
#pragma unroll
  for (int q = 0; q < 4; ++q) {
    float4 f = *(const float4*)(xr + lane * 16 + q * 4);
    v[q*4+0] = f.x; v[q*4+1] = f.y; v[q*4+2] = f.z; v[q*4+3] = f.w;
  }
#pragma unroll
  for (int j = 0; j < 16; ++j) sum += v[j];
#pragma unroll
  for (int o = 16; o >= 1; o >>= 1) sum += __shfl_xor(sum, o);
  float mu = sum * (1.f / D_);
  float sq = 0.f;
#pragma unroll
  for (int j = 0; j < 16; ++j) { float d = v[j] - mu; sq += d * d; }
#pragma unroll
  for (int o = 16; o >= 1; o >>= 1) sq += __shfl_xor(sq, o);
  float rstd = rsqrtf(sq * (1.f / D_) + 1e-5f);

  union { bf16 hb[16]; uint4 u[2]; } hh;
#pragma unroll
  for (int j = 0; j < 16; ++j) {
    int d = lane * 16 + j;
    float t = (v[j] - mu) * rstd * g[d] + b[d];
    v[j] = t;
    hh.hb[j] = (bf16)t;
  }
  uint4* hp = (uint4*)(h + (size_t)token * D_ + lane * 16);
  hp[0] = hh.u[0]; hp[1] = hh.u[1];

  // router logits: 16 dot products of length 512 (reduced across the wave)
  float logit[16];
#pragma unroll
  for (int e2 = 0; e2 < 16; ++e2) {
    float p = 0.f;
#pragma unroll
    for (int j = 0; j < 16; ++j) p += v[j] * rw[(lane * 16 + j) * EXP_ + e2];
#pragma unroll
    for (int o = 16; o >= 1; o >>= 1) p += __shfl_xor(p, o);
    logit[e2] = p + rb[e2];
  }
  // softmax(logits / 0.7)
  float mx = -1e30f;
#pragma unroll
  for (int e2 = 0; e2 < 16; ++e2) mx = fmaxf(mx, logit[e2]);
  float se = 0.f, pr[16];
#pragma unroll
  for (int e2 = 0; e2 < 16; ++e2) { pr[e2] = __expf((logit[e2] - mx) * (1.f / 0.7f)); se += pr[e2]; }
  float inv = 1.f / se;
  int i1 = 0; float v1 = -1.f;
#pragma unroll
  for (int e2 = 0; e2 < 16; ++e2) if (pr[e2] > v1) { v1 = pr[e2]; i1 = e2; }
  int i2 = 0; float v2 = -1.f;
#pragma unroll
  for (int e2 = 0; e2 < 16; ++e2) if (e2 != i1 && pr[e2] > v2) { v2 = pr[e2]; i2 = e2; }
  if (lane == 0) {
    int p1 = atomicAdd(&counts[i1], 1);
    toklist[i1 * NTOK + p1] = token * 2 + 0;
    wlist  [i1 * NTOK + p1] = v1 * inv;
    int p2 = atomicAdd(&counts[i2], 1);
    toklist[i2 * NTOK + p2] = token * 2 + 1;
    wlist  [i2 * NTOK + p2] = v2 * inv;
  }
}

// ---------------- combine expert slots + shared, second LayerNorm ----------------
__global__ __launch_bounds__(256) void ln_combine_k(
    const float* __restrict__ ybuf, const float* __restrict__ shbuf,
    const float* __restrict__ g, const float* __restrict__ b,
    float* __restrict__ moe, bf16* __restrict__ fh)
{
  int token = blockIdx.x * 8 + (threadIdx.x >> 5);
  int lane  = threadIdx.x & 31;
  int d0 = lane * 16;
  float v[16];
  float sum = 0.f;
#pragma unroll
  for (int q = 0; q < 4; ++q) {
    float4 a = *(const float4*)(ybuf + (size_t)(token * 2 + 0) * D_ + d0 + q * 4);
    float4 c = *(const float4*)(ybuf + (size_t)(token * 2 + 1) * D_ + d0 + q * 4);
    float4 s = *(const float4*)(shbuf + (size_t)token * D_ + d0 + q * 4);
    v[q*4+0] = a.x + c.x + s.x; v[q*4+1] = a.y + c.y + s.y;
    v[q*4+2] = a.z + c.z + s.z; v[q*4+3] = a.w + c.w + s.w;
  }
#pragma unroll
  for (int j = 0; j < 16; ++j) sum += v[j];
#pragma unroll
  for (int o = 16; o >= 1; o >>= 1) sum += __shfl_xor(sum, o);
  float mu = sum * (1.f / D_);
  float sq = 0.f;
#pragma unroll
  for (int j = 0; j < 16; ++j) { float d = v[j] - mu; sq += d * d; }
#pragma unroll
  for (int o = 16; o >= 1; o >>= 1) sq += __shfl_xor(sq, o);
  float rstd = rsqrtf(sq * (1.f / D_) + 1e-5f);

  union { bf16 hb[16]; uint4 u[2]; } hh;
  float* mp = moe + (size_t)token * D_ + d0;
#pragma unroll
  for (int j = 0; j < 16; ++j) {
    mp[j] = v[j];
    hh.hb[j] = (bf16)((v[j] - mu) * rstd * g[d0 + j] + b[d0 + j]);
  }
  uint4* fp = (uint4*)(fh + (size_t)token * D_ + d0);
  fp[0] = hh.u[0]; fp[1] = hh.u[1];
}

// ---------------- grouped bf16 WMMA GEMM ----------------
// C[M,N] = A[M,K] * B[K,N], B stored transposed as BT[N,K] (bf16).
// Macro tile 128(M) x 256(N), 8 waves as 2(M) x 4(N), wave tile 64x64.
// EPI: 0 = silu -> bf16 out (row offset offs[e] when GATHER)
//      1 = 0.25*(c+bias) -> f32
//      2 = (c+bias)*w[row] scattered to f32 ybuf[tokslot]
//      3 = res1 + res2 + (c+bias) -> f32 (final output)
#define LDSTRIDE 40
template<int EPI, bool GATHER>
__global__ __launch_bounds__(256) void gemm_bf16_k(
    const bf16* __restrict__ Abase, const bf16* __restrict__ BTbase,
    const float* __restrict__ biasBase,
    bf16* __restrict__ outB, float* __restrict__ outF,
    const float* __restrict__ res1, const float* __restrict__ res2,
    const int* __restrict__ tokAll, const float* __restrict__ wAll,
    const int* __restrict__ counts, const int* __restrict__ offs,
    int M, int N, int K)
{
  const int tid  = threadIdx.x;
  const int lane = tid & 31;
  const int wid  = tid >> 5;
  const int wmO  = (wid >> 2) * 64;   // wave M offset
  const int wnO  = (wid & 3) * 64;    // wave N offset
  const int ex   = blockIdx.z;
  const int nTile = blockIdx.x * 256;
  const int mTile = blockIdx.y * 128;

  const bf16*  A    = Abase;
  const bf16*  BT   = BTbase + (size_t)ex * (size_t)N * (size_t)K;
  const float* bias = biasBase + (size_t)ex * (size_t)N;
  const int*   tok  = nullptr;
  const float* wv   = nullptr;
  int Meff = M, rowOff = 0;
  if (GATHER || EPI == 2) {
    Meff = counts[ex];
    tok  = tokAll + ex * NTOK;
    if (EPI == 2) { wv = wAll + ex * NTOK; A = Abase + (size_t)offs[ex] * (size_t)K; }
    else rowOff = offs[ex];
  }
  if (mTile >= Meff) return;   // uniform early-exit for empty tiles

  __shared__ bf16 sA[2][128 * LDSTRIDE];
  __shared__ bf16 sB[2][256 * LDSTRIDE];
  __shared__ int  sTok[128];
  if (GATHER) {
    if (tid < 128) {
      int r = mTile + tid; if (r >= Meff) r = Meff - 1;
      sTok[tid] = tok[r];
    }
    __syncthreads();
  }

  auto stage = [&](int buf, int kt) {
#pragma unroll
    for (int i = 0; i < 2; ++i) {         // A tile 128x32 (512 chunks of 16B)
      int lin = tid + i * 256;
      int r = lin >> 2, c4 = (lin & 3) * 8;
      const bf16* gp;
      if (GATHER) {
        int token = sTok[r] >> 1;
        gp = A + (size_t)token * K + kt + c4;
      } else {
        int rr = mTile + r; if (rr >= Meff) rr = Meff - 1;
        gp = A + (size_t)rr * K + kt + c4;
      }
      async_b128(&sA[buf][r * LDSTRIDE + c4], gp);
    }
#pragma unroll
    for (int i = 0; i < 4; ++i) {         // BT tile 256x32 (1024 chunks of 16B)
      int lin = tid + i * 256;
      int r = lin >> 2, c4 = (lin & 3) * 8;
      async_b128(&sB[buf][r * LDSTRIDE + c4], BT + (size_t)(nTile + r) * K + kt + c4);
    }
  };

  v8f acc[4][4];
#pragma unroll
  for (int a = 0; a < 4; ++a)
#pragma unroll
    for (int c = 0; c < 4; ++c)
      acc[a][c] = (v8f){0.f,0.f,0.f,0.f,0.f,0.f,0.f,0.f};

  stage(0, 0);
  const int nk = K / 32;
  const int kh = (lane >> 4) * 8;
  for (int t = 0; t < nk; ++t) {
    int buf = t & 1;
    if (t + 1 < nk) { stage(buf ^ 1, (t + 1) * 32); wait_async_le6(); }
    else            { wait_async_0(); }
    __syncthreads();

    const bf16* pA = &sA[buf][0];
    const bf16* pB = &sB[buf][0];
    v16bf bfr[4], afr[4];
#pragma unroll
    for (int fn = 0; fn < 4; ++fn)
      bfr[fn] = frag_ld(pB + (wnO + fn * 16 + (lane & 15)) * LDSTRIDE + kh);
#pragma unroll
    for (int fm = 0; fm < 4; ++fm)
      afr[fm] = frag_ld(pA + (wmO + fm * 16 + (lane & 15)) * LDSTRIDE + kh);
#pragma unroll
    for (int fm = 0; fm < 4; ++fm)
#pragma unroll
      for (int fn = 0; fn < 4; ++fn)
        acc[fm][fn] = __builtin_amdgcn_wmma_f32_16x16x32_bf16(
            false, afr[fm], false, bfr[fn], (short)0, acc[fm][fn], false, false);
    __syncthreads();
  }

  // epilogue: C element (m,n): vgpr v, lane L -> m = 8*(L>>4)+v, n = L&15
#pragma unroll
  for (int fm = 0; fm < 4; ++fm) {
    int rB = mTile + wmO + fm * 16 + ((lane >> 4) * 8);
#pragma unroll
    for (int fn = 0; fn < 4; ++fn) {
      int col = nTile + wnO + fn * 16 + (lane & 15);
      float bi = bias[col];
#pragma unroll
      for (int v = 0; v < 8; ++v) {
        int row = rB + v;
        float val = acc[fm][fn][v] + bi;
        if (EPI == 0) {
          if (!GATHER || row < Meff) {
            float s = val / (1.f + __expf(-val));
            outB[(size_t)(rowOff + row) * N + col] = (bf16)s;
          }
        } else if (EPI == 1) {
          outF[(size_t)row * N + col] = 0.25f * val;
        } else if (EPI == 2) {
          if (row < Meff) {
            int ts = tok[row];
            outF[(size_t)ts * N + col] = val * wv[row];
          }
        } else {
          size_t o = (size_t)row * N + col;
          outF[o] = res1[o] + res2[o] + val;
        }
      }
    }
  }
}

// ---------------- host launcher ----------------
extern "C" void kernel_launch(void* const* d_in, const int* in_sizes, int n_in,
                              void* d_out, int out_size, void* d_ws, size_t ws_size,
                              hipStream_t stream) {
  (void)in_sizes; (void)n_in; (void)out_size; (void)ws_size;
  const float* x    = (const float*)d_in[0];
  const float* lng  = (const float*)d_in[1];
  const float* lnb  = (const float*)d_in[2];
  const float* rw   = (const float*)d_in[3];
  const float* rb   = (const float*)d_in[4];
  const float* ew1  = (const float*)d_in[5];
  const float* eb1  = (const float*)d_in[6];
  const float* ew2  = (const float*)d_in[7];
  const float* eb2  = (const float*)d_in[8];
  const float* sw1  = (const float*)d_in[9];
  const float* sb1  = (const float*)d_in[10];
  const float* sw2  = (const float*)d_in[11];
  const float* sb2  = (const float*)d_in[12];
  const float* flg  = (const float*)d_in[13];
  const float* flb  = (const float*)d_in[14];
  const float* fw1  = (const float*)d_in[15];
  const float* fb1  = (const float*)d_in[16];
  const float* fw2  = (const float*)d_in[17];
  const float* fb2  = (const float*)d_in[18];
  float* out = (float*)d_out;

  char* w = (char*)d_ws;
  size_t off = 0;
  auto alloc = [&](size_t bytes) -> void* {
    void* p = w + off; off += bytes; off = (off + 255) & ~(size_t)255; return p;
  };
  bf16* ew1T = (bf16*)alloc((size_t)EXP_ * DH_ * D_ * 2);
  bf16* ew2T = (bf16*)alloc((size_t)EXP_ * D_ * DH_ * 2);
  bf16* sw1T = (bf16*)alloc((size_t)DS2_ * D_ * 2);
  bf16* sw2T = (bf16*)alloc((size_t)D_ * DS2_ * 2);
  bf16* fw1T = (bf16*)alloc((size_t)DH_ * D_ * 2);
  bf16* fw2T = (bf16*)alloc((size_t)D_ * DH_ * 2);
  bf16* hbuf = (bf16*)alloc((size_t)NTOK * D_ * 2);
  bf16* A1   = (bf16*)alloc((size_t)NTOK * DS2_ * 2);
  float* shb = (float*)alloc((size_t)NTOK * D_ * 4);
  bf16* G1   = (bf16*)alloc((size_t)NTOK * 2 * DH_ * 2);
  bf16* F1   = (bf16*)alloc((size_t)NTOK * DH_ * 2);
  float* yb  = (float*)alloc((size_t)NTOK * 2 * D_ * 4);
  float* moe = (float*)alloc((size_t)NTOK * D_ * 4);
  bf16* fh   = (bf16*)alloc((size_t)NTOK * D_ * 2);
  int*  tokl = (int*)alloc((size_t)EXP_ * NTOK * 4);
  float* wl  = (float*)alloc((size_t)EXP_ * NTOK * 4);
  int*  cnt  = (int*)alloc(64);
  int*  offs = (int*)alloc(64);

  zero_counts_k<<<1, 32, 0, stream>>>(cnt);

  // weight transposes (fp32 -> bf16 [N,K])
  convT_k<<<dim3(4096, 1, EXP_), 256, 0, stream>>>(ew1, ew1T, D_, DH_);
  convT_k<<<dim3(4096, 1, EXP_), 256, 0, stream>>>(ew2, ew2T, DH_, D_);
  convT_k<<<dim3(2048, 1, 1),    256, 0, stream>>>(sw1, sw1T, D_, DS2_);
  convT_k<<<dim3(2048, 1, 1),    256, 0, stream>>>(sw2, sw2T, DS2_, D_);
  convT_k<<<dim3(4096, 1, 1),    256, 0, stream>>>(fw1, fw1T, D_, DH_);
  convT_k<<<dim3(4096, 1, 1),    256, 0, stream>>>(fw2, fw2T, DH_, D_);

  // LayerNorm + router + top-2 dispatch
  ln_router_k<<<dim3(NTOK / 8), 256, 0, stream>>>(x, lng, lnb, rw, rb, hbuf, tokl, wl, cnt);
  prefix16_k<<<1, 32, 0, stream>>>(cnt, offs);

  // shared expert
  gemm_bf16_k<0, false><<<dim3(DS2_ / 256, NTOK / 128, 1), 256, 0, stream>>>(
      hbuf, sw1T, sb1, A1, nullptr, nullptr, nullptr, nullptr, nullptr, nullptr, nullptr,
      NTOK, DS2_, D_);
  gemm_bf16_k<1, false><<<dim3(D_ / 256, NTOK / 128, 1), 256, 0, stream>>>(
      A1, sw2T, sb2, nullptr, shb, nullptr, nullptr, nullptr, nullptr, nullptr, nullptr,
      NTOK, D_, DS2_);

  // expert FFNs (grouped, gathered rows; early-exit on per-expert count)
  gemm_bf16_k<0, true><<<dim3(DH_ / 256, NTOK / 128, EXP_), 256, 0, stream>>>(
      hbuf, ew1T, eb1, G1, nullptr, nullptr, nullptr, tokl, nullptr, cnt, offs,
      NTOK, DH_, D_);
  gemm_bf16_k<2, false><<<dim3(D_ / 256, NTOK / 128, EXP_), 256, 0, stream>>>(
      G1, ew2T, eb2, nullptr, yb, nullptr, nullptr, tokl, wl, cnt, offs,
      NTOK, D_, DH_);

  // combine + LN
  ln_combine_k<<<dim3(NTOK / 8), 256, 0, stream>>>(yb, shb, flg, flb, moe, fh);

  // final FF with fused residual epilogue
  gemm_bf16_k<0, false><<<dim3(DH_ / 256, NTOK / 128, 1), 256, 0, stream>>>(
      fh, fw1T, fb1, F1, nullptr, nullptr, nullptr, nullptr, nullptr, nullptr, nullptr,
      NTOK, DH_, D_);
  gemm_bf16_k<3, false><<<dim3(D_ / 256, NTOK / 128, 1), 256, 0, stream>>>(
      F1, fw2T, fb2, nullptr, out, x, moe, nullptr, nullptr, nullptr, nullptr,
      NTOK, D_, DH_);
}
